// OdorPredictorV6_10771777978577
// MI455X (gfx1250) — matrix operand order repaired and checked by reading, not compiled, not run on hardware
//
#include <hip/hip_runtime.h>
#include <hip/hip_bf16.h>
#include <math.h>

// ---------------------------------------------------------------------------
// CDNA5 (gfx1250) fused SchNet-style CFConv layer.
// All GEMMs run on v_wmma_f32_16x16x32_bf16 (wave32, 16x16 tiles, f32 accum).
// Edge pipeline blocks 64 edges (4 M-tiles) so each 32B B-fragment load from
// the L2-resident packed weights feeds 4 WMMAs (register reuse).
// ---------------------------------------------------------------------------

typedef __bf16 bf16_t;
typedef __attribute__((ext_vector_type(16))) __bf16 v16bf;
typedef __attribute__((ext_vector_type(8)))  float  v8f;

#define HDIM 256
#define RDIM 64
#define CUTOFF_F 10.0f
#define EPS_W 1e-5f
#define EPS_LN 1e-5f
#define PI_F 3.14159265358979323846f

// bf16 elements per swizzled M-tile activation buffer (8 k-steps of 32).
#define A_STRIDE (8 * 32 * 16)

__device__ __forceinline__ float sigmoidf_(float v) { return 1.f / (1.f + __expf(-v)); }

__device__ __forceinline__ v8f vzero8() {
  v8f z = {0.f, 0.f, 0.f, 0.f, 0.f, 0.f, 0.f, 0.f};
  return z;
}

// ---------------------------------------------------------------------------
// A-fragment swizzle for 16-bit A (16x32 per k-step), per ISA 7.12.2:
//   lane L: M = L%16, half h = L/16; h=0 covers K in {0..7,16..23}, h=1 rest.
//   element i: K = (i<8) ? (8h+i) : (16+8h+(i-8))
// LDS layout A_sw[ktile][lane(32)][i(16)] -> each lane reads 32B contiguous.
// Inverse (k within 32 -> (h,i)): h=(kk>>3)&1 ; i=(kk&16)?8+(kk&7):(kk&7)
// ---------------------------------------------------------------------------
__device__ __forceinline__ int a_swz_index(int ktile, int m, int kk) {
  int hh = (kk & 8) ? 16 : 0;
  int ii = (kk & 16) ? (8 + (kk & 7)) : (kk & 7);
  return ((ktile * 32 + hh + m) * 16) + ii;
}

// Multi-M WMMA panel: one 16-col output panel for MT stacked 16-row A tiles.
// B fragment is loaded once per k-step and reused for all MT WMMAs.
// Bp pre-packed: [ntile][ktile][lane(32)][i(16)] contiguous bf16.
template <int MT>
__device__ __forceinline__ void wmma_panel_mt(const bf16_t* A_sw, const bf16_t* Bp,
                                              int ntile, int ktiles, v8f* acc) {
  const int lane = threadIdx.x & 31;
  const bf16_t* ab = A_sw + lane * 16;
  const bf16_t* bb = Bp + ((size_t)(ntile * ktiles) * 32 + lane) * 16;
  for (int kt = 0; kt < ktiles; ++kt) {
    v16bf b = *(const v16bf*)(bb + (size_t)kt * 32 * 16);
#pragma unroll
    for (int mt = 0; mt < MT; ++mt) {
      v16bf a = *(const v16bf*)(ab + mt * A_STRIDE + kt * 32 * 16);
      acc[mt] = __builtin_amdgcn_wmma_f32_16x16x32_bf16(false, a, false, b,
                                                        (short)0, acc[mt], false, false);
    }
  }
}

// Epilogue: C/D layout — lane L: N = ntile*16 + L%16; vgpr r: M = r + 8*(L/16).
// Adds bias, optional SiLU, writes bf16 back into the A-swizzle for next GEMM.
__device__ __forceinline__ void epilogue_swz(bf16_t* dst, v8f c, int ntile,
                                             const float* bias, bool do_silu) {
  const int lane = threadIdx.x & 31;
  const int nn = lane & 15;
  const int n = ntile * 16 + nn;
  const float bv = bias[n];
  const int kt = n >> 5, kk = n & 31;
  const int mbase = (lane >> 4) * 8;
  const int hh = (kk & 8) ? 16 : 0;
  const int ii = (kk & 16) ? (8 + (kk & 7)) : (kk & 7);
#pragma unroll
  for (int r = 0; r < 8; ++r) {
    float v = c[r] + bv;
    if (do_silu) v = v * sigmoidf_(v);
    dst[((kt * 32 + hh + (mbase + r)) * 16) + ii] = (bf16_t)v;
  }
}

// ---------------------------------------------------------------------------
// Kernel 0: pack row-major f32 weight [K,N] into bf16 WMMA B-operand layout:
//   B fragment (16-bit B, 32x16/k-step): lane = ((kk&8)?16:0)+nn, element i=(kk&16)?8+(kk&7):(kk&7).
// ---------------------------------------------------------------------------
__global__ void pack_w_kernel(const float* __restrict__ src, bf16_t* __restrict__ dst,
                              int K, int N) {
  int e = blockIdx.x * blockDim.x + threadIdx.x;
  if (e >= K * N) return;
  int k = e / N, n = e - k * N;
  int ktiles = K >> 5;
  int kt = k >> 5, kk = k & 31;
  int nt = n >> 4, nn = n & 15;
  int lane = ((kk & 8) ? 16 : 0) + nn;
  int i = (kk & 16) ? (8 + (kk & 7)) : (kk & 7);
  dst[(((size_t)(nt * ktiles + kt) * 32 + lane) * 16) + i] = (bf16_t)src[e];
}

// ---------------------------------------------------------------------------
// Kernel 1: y = x @ node_w + node_b (hoisted out of the per-edge loop).
// ---------------------------------------------------------------------------
__global__ void __launch_bounds__(256) node_pre_kernel(
    const float* __restrict__ x, const bf16_t* __restrict__ p_node_w,
    const float* __restrict__ node_b, float* __restrict__ y, int N) {
  __shared__ __align__(32) bf16_t bufA[A_STRIDE];
  const int t = threadIdx.x, wave = t >> 5, lane = t & 31;
  const int n0 = blockIdx.x * 16;

  {  // stage x tile [16,256] -> swizzled bf16
    const int m = t >> 4;
    const int c0 = (t & 15) * 16;
    const float* xr = x + (size_t)(n0 + m) * HDIM + c0;
#pragma unroll
    for (int j = 0; j < 16; ++j) {
      int k = c0 + j;
      bufA[a_swz_index(k >> 5, m, k & 31)] = (bf16_t)xr[j];
    }
  }
  __syncthreads();

  const int nn = lane & 15, mbase = (lane >> 4) * 8;
  v8f ca = vzero8(), cb = vzero8();
  wmma_panel_mt<1>(bufA, p_node_w, wave, 8, &ca);
  wmma_panel_mt<1>(bufA, p_node_w, wave + 8, 8, &cb);
  const int na = wave * 16 + nn, nb = (wave + 8) * 16 + nn;
  const float ba = node_b[na], bb = node_b[nb];
#pragma unroll
  for (int r = 0; r < 8; ++r) {
    y[(size_t)(n0 + mbase + r) * HDIM + na] = ca[r] + ba;
    y[(size_t)(n0 + mbase + r) * HDIM + nb] = cb[r] + bb;
  }
}

// ---------------------------------------------------------------------------
// Kernel 2: fused per-edge pipeline, 64 edges (4 M-tiles) per 256-thread block:
//   rbf -> GEMM(64x64)+silu -> GEMM(64x64) -> GEMM(64x256)+silu
//       -> GEMM(256x256)+silu -> GEMM(256x256)=W -> agg[row] += y[col]*W
// Each B fragment (32B/lane) feeds MT WMMAs from registers.
// ---------------------------------------------------------------------------
__global__ void __launch_bounds__(256) edge_kernel(
    const int* __restrict__ ei, const float* __restrict__ dist,
    const float* __restrict__ centers, const float* __restrict__ widths,
    const bf16_t* __restrict__ p_rbf_w1, const float* __restrict__ rbf_b1,
    const bf16_t* __restrict__ p_rbf_w2, const float* __restrict__ rbf_b2,
    const bf16_t* __restrict__ p_filt_w1, const float* __restrict__ filt_b1,
    const bf16_t* __restrict__ p_filt_w2, const float* __restrict__ filt_b2,
    const bf16_t* __restrict__ p_filt_w3, const float* __restrict__ filt_b3,
    const float* __restrict__ y, float* __restrict__ agg, int E) {
  __shared__ __align__(32) bf16_t bufA[4 * A_STRIDE];  // 32 KB
  __shared__ __align__(32) bf16_t bufB[4 * A_STRIDE];  // 32 KB
  __shared__ int rowsh[64];
  __shared__ int colsh[64];

  const int t = threadIdx.x, wave = t >> 5, lane = t & 31;
  const int e0 = blockIdx.x * 64;

  if (t < 64) {
    int e = e0 + t;
    int ec = (e < E) ? e : 0;
    rowsh[t] = ei[ec];
    colsh[t] = ei[E + ec];
  }

  // Hint-prefetch the hot L2-resident weight panels (global_prefetch_b8).
  __builtin_prefetch(p_filt_w2 + (t << 8), 0, 0);
  __builtin_prefetch(p_filt_w3 + (t << 8), 0, 0);

  // Stage 0: RBF features [64 edges x 64 basis] -> bufA (swizzled bf16).
  // Thread t: edge m = t/4, basis chunk (t%4)*16 .. +15 (d/env computed once).
  {
    const int m = t >> 2;
    const int jb = (t & 3) * 16;
    const int e = e0 + m;
    const float d = dist[(e < E) ? e : (E - 1)];
    const float env = (d < CUTOFF_F) ? 0.5f * (__cosf(d * (PI_F / CUTOFF_F)) + 1.f) : 0.f;
    const int mt = m >> 4, mm = m & 15;
    bf16_t* dst = bufA + mt * A_STRIDE;
#pragma unroll
    for (int u = 0; u < 16; ++u) {
      int j = jb + u;
      float w = fabsf(widths[j]) + EPS_W;
      float z = (d - centers[j]) / w;
      float val = __expf(-0.5f * z * z) * env;
      dst[a_swz_index(j >> 5, mm, j & 31)] = (bf16_t)val;
    }
  }
  __syncthreads();

  // GEMM1: silu(rbf @ rbf_w1 + b1). 4 ntiles x 4 mtiles over 8 waves:
  // wave handles ntile = wave&3, mtile pair = (wave>>2)*2.
  {
    const int nt = wave & 3;
    const int mb = (wave >> 2) * 2;
    v8f acc[2] = {vzero8(), vzero8()};
    wmma_panel_mt<2>(bufA + mb * A_STRIDE, p_rbf_w1, nt, 2, acc);
    epilogue_swz(bufB + (mb + 0) * A_STRIDE, acc[0], nt, rbf_b1, true);
    epilogue_swz(bufB + (mb + 1) * A_STRIDE, acc[1], nt, rbf_b1, true);
  }
  __syncthreads();
  // GEMM2: (.. @ rbf_w2 + b2)
  {
    const int nt = wave & 3;
    const int mb = (wave >> 2) * 2;
    v8f acc[2] = {vzero8(), vzero8()};
    wmma_panel_mt<2>(bufB + mb * A_STRIDE, p_rbf_w2, nt, 2, acc);
    epilogue_swz(bufA + (mb + 0) * A_STRIDE, acc[0], nt, rbf_b2, false);
    epilogue_swz(bufA + (mb + 1) * A_STRIDE, acc[1], nt, rbf_b2, false);
  }
  __syncthreads();
  // GEMM3: silu(r @ filt_w1 + b)  -> bufB (16 ntiles, 2 per wave, MT=4)
  for (int s = 0; s < 2; ++s) {
    const int nt = wave + 8 * s;
    v8f acc[4] = {vzero8(), vzero8(), vzero8(), vzero8()};
    wmma_panel_mt<4>(bufA, p_filt_w1, nt, 2, acc);
#pragma unroll
    for (int mt = 0; mt < 4; ++mt)
      epilogue_swz(bufB + mt * A_STRIDE, acc[mt], nt, filt_b1, true);
  }
  __syncthreads();
  // GEMM4: silu(.. @ filt_w2 + b) -> bufA
  for (int s = 0; s < 2; ++s) {
    const int nt = wave + 8 * s;
    v8f acc[4] = {vzero8(), vzero8(), vzero8(), vzero8()};
    wmma_panel_mt<4>(bufB, p_filt_w2, nt, 8, acc);
#pragma unroll
    for (int mt = 0; mt < 4; ++mt)
      epilogue_swz(bufA + mt * A_STRIDE, acc[mt], nt, filt_b2, true);
  }
  __syncthreads();
  // GEMM5: W = .. @ filt_w3 + b, fused with gather-modulate-scatter:
  //   agg[row[m], n] += (W[m,n]) * y[col[m], n]
  // For fixed (mt,r): lanes 0..15 touch 16 consecutive floats (coalesced 64B).
  {
    const int nn = lane & 15, mbase = (lane >> 4) * 8;
    for (int s = 0; s < 2; ++s) {
      const int nt = wave + 8 * s;
      const int n = nt * 16 + nn;
      const float bv = filt_b3[n];
      v8f acc[4] = {vzero8(), vzero8(), vzero8(), vzero8()};
      wmma_panel_mt<4>(bufA, p_filt_w3, nt, 8, acc);
#pragma unroll
      for (int mt = 0; mt < 4; ++mt) {
#pragma unroll
        for (int r = 0; r < 8; ++r) {
          const int m = mt * 16 + mbase + r;
          const int e = e0 + m;
          if (e < E) {
            const float wv = acc[mt][r] + bv;
            const int row = rowsh[m], col = colsh[m];
            const float msg = wv * y[(size_t)col * HDIM + n];
            atomicAdd(agg + (size_t)row * HDIM + n, msg);
          }
        }
      }
    }
  }
}

// ---------------------------------------------------------------------------
// Kernel 3: h = x + agg; LayerNorm; out = ln * sigmoid(ln @ gate_w + gate_b)
// ---------------------------------------------------------------------------
__global__ void __launch_bounds__(256) node_post_kernel(
    const float* __restrict__ x, const float* __restrict__ agg,
    const float* __restrict__ ln_g, const float* __restrict__ ln_b,
    const bf16_t* __restrict__ p_gate_w, const float* __restrict__ gate_b,
    float* __restrict__ out, int N) {
  __shared__ __align__(32) bf16_t bufA[A_STRIDE];
  __shared__ float lnbuf[16 * HDIM];
  __shared__ float rsum[256];
  __shared__ float rsq[256];

  const int t = threadIdx.x, wave = t >> 5, lane = t & 31;
  const int n0 = blockIdx.x * 16;
  const int m = t >> 4;
  const int c0 = (t & 15) * 16;

  float hreg[16];
  float s = 0.f, q = 0.f;
  const float* xr = x + (size_t)(n0 + m) * HDIM + c0;
  const float* ar = agg + (size_t)(n0 + m) * HDIM + c0;
#pragma unroll
  for (int j = 0; j < 16; ++j) {
    float v = xr[j] + ar[j];
    hreg[j] = v;
    s += v;
    q += v * v;
  }
  rsum[t] = s;
  rsq[t] = q;
  __syncthreads();
  float S = 0.f, Q = 0.f;
#pragma unroll
  for (int j = 0; j < 16; ++j) {
    S += rsum[m * 16 + j];
    Q += rsq[m * 16 + j];
  }
  const float mu = S * (1.f / HDIM);
  const float var = Q * (1.f / HDIM) - mu * mu;
  const float inv = rsqrtf(var + EPS_LN);
#pragma unroll
  for (int j = 0; j < 16; ++j) {
    int k = c0 + j;
    float v = (hreg[j] - mu) * inv * ln_g[k] + ln_b[k];
    lnbuf[m * HDIM + k] = v;
    bufA[a_swz_index(k >> 5, m, k & 31)] = (bf16_t)v;
  }
  __syncthreads();

  const int nn = lane & 15, mbase = (lane >> 4) * 8;
  v8f ca = vzero8(), cb = vzero8();
  wmma_panel_mt<1>(bufA, p_gate_w, wave, 8, &ca);
  wmma_panel_mt<1>(bufA, p_gate_w, wave + 8, 8, &cb);
  const int na = wave * 16 + nn, nb = (wave + 8) * 16 + nn;
  const float ba = gate_b[na], bb = gate_b[nb];
#pragma unroll
  for (int r = 0; r < 8; ++r) {
    int mm = mbase + r;
    out[(size_t)(n0 + mm) * HDIM + na] = lnbuf[mm * HDIM + na] * sigmoidf_(ca[r] + ba);
    out[(size_t)(n0 + mm) * HDIM + nb] = lnbuf[mm * HDIM + nb] * sigmoidf_(cb[r] + bb);
  }
}

// ---------------------------------------------------------------------------
extern "C" void kernel_launch(void* const* d_in, const int* in_sizes, int n_in,
                              void* d_out, int out_size, void* d_ws, size_t ws_size,
                              hipStream_t stream) {
  const float* x       = (const float*)d_in[0];
  const int*   ei      = (const int*)d_in[1];
  const float* dist    = (const float*)d_in[2];
  const float* centers = (const float*)d_in[3];
  const float* widths  = (const float*)d_in[4];
  const float* rbf_w1  = (const float*)d_in[5];
  const float* rbf_b1  = (const float*)d_in[6];
  const float* rbf_w2  = (const float*)d_in[7];
  const float* rbf_b2  = (const float*)d_in[8];
  const float* filt_w1 = (const float*)d_in[9];
  const float* filt_b1 = (const float*)d_in[10];
  const float* filt_w2 = (const float*)d_in[11];
  const float* filt_b2 = (const float*)d_in[12];
  const float* filt_w3 = (const float*)d_in[13];
  const float* filt_b3 = (const float*)d_in[14];
  const float* node_w  = (const float*)d_in[15];
  const float* node_b  = (const float*)d_in[16];
  const float* ln_g    = (const float*)d_in[17];
  const float* ln_b    = (const float*)d_in[18];
  const float* gate_w  = (const float*)d_in[19];
  const float* gate_b  = (const float*)d_in[20];

  const int N = in_sizes[0] / HDIM;
  const int E = in_sizes[2];

  // Workspace carve-up (256B aligned).
  char* ws = (char*)d_ws;
  size_t off = 0;
  auto carve = [&](size_t bytes) -> void* {
    void* p = ws + off;
    off = (off + bytes + 255) & ~(size_t)255;
    return p;
  };
  float*  y        = (float*)carve((size_t)N * HDIM * sizeof(float));
  float*  agg      = (float*)carve((size_t)N * HDIM * sizeof(float));
  bf16_t* p_rbf_w1 = (bf16_t*)carve((size_t)RDIM * RDIM * sizeof(bf16_t));
  bf16_t* p_rbf_w2 = (bf16_t*)carve((size_t)RDIM * RDIM * sizeof(bf16_t));
  bf16_t* p_filt_w1= (bf16_t*)carve((size_t)RDIM * HDIM * sizeof(bf16_t));
  bf16_t* p_filt_w2= (bf16_t*)carve((size_t)HDIM * HDIM * sizeof(bf16_t));
  bf16_t* p_filt_w3= (bf16_t*)carve((size_t)HDIM * HDIM * sizeof(bf16_t));
  bf16_t* p_node_w = (bf16_t*)carve((size_t)HDIM * HDIM * sizeof(bf16_t));
  bf16_t* p_gate_w = (bf16_t*)carve((size_t)HDIM * HDIM * sizeof(bf16_t));

  hipMemsetAsync(agg, 0, (size_t)N * HDIM * sizeof(float), stream);

  // Pack weights into WMMA B-operand bf16 layout.
  auto packs = [&](const float* s, bf16_t* d, int K, int Nn) {
    int total = K * Nn;
    pack_w_kernel<<<(total + 255) / 256, 256, 0, stream>>>(s, d, K, Nn);
  };
  packs(rbf_w1, p_rbf_w1, RDIM, RDIM);
  packs(rbf_w2, p_rbf_w2, RDIM, RDIM);
  packs(filt_w1, p_filt_w1, RDIM, HDIM);
  packs(filt_w2, p_filt_w2, HDIM, HDIM);
  packs(filt_w3, p_filt_w3, HDIM, HDIM);
  packs(node_w, p_node_w, HDIM, HDIM);
  packs(gate_w, p_gate_w, HDIM, HDIM);

  // y = x @ node_w + node_b
  node_pre_kernel<<<N / 16, 256, 0, stream>>>(x, p_node_w, node_b, y, N);

  // Fused per-edge filter pipeline + scatter-add (64 edges / block).
  edge_kernel<<<(E + 63) / 64, 256, 0, stream>>>(
      ei, dist, centers, widths,
      p_rbf_w1, rbf_b1, p_rbf_w2, rbf_b2,
      p_filt_w1, filt_b1, p_filt_w2, filt_b2, p_filt_w3, filt_b3,
      y, agg, E);

  // LayerNorm + gated output.
  node_post_kernel<<<N / 16, 256, 0, stream>>>(
      x, agg, ln_g, ln_b, p_gate_w, gate_b, (float*)d_out, N);
}
